// MultiHeadAttention_13864154432374
// MI455X (gfx1250) — matrix operand order repaired and checked by reading, not compiled, not run on hardware
//
#include <hip/hip_runtime.h>
#include <hip/hip_bf16.h>
#include <math.h>

typedef __attribute__((ext_vector_type(16))) _Float16 v16h;
typedef __attribute__((ext_vector_type(8)))  float    v8f;

#define BB 4
#define LL 2048
#define DD 256
#define HH 8
#define HD 32

union V16H {
  v16h v;
  _Float16 h[16];
  unsigned int u[8];
};
union PKH {
  unsigned int u;
  _Float16 f[2];
};

// ---- lane <-> lane^16 swap (half-wave exchange), VALU if permlanex16 exists ----
__device__ __forceinline__ int swap16i(int x) {
#if defined(__has_builtin) && __has_builtin(__builtin_amdgcn_permlanex16)
  return __builtin_amdgcn_permlanex16(x, x, 0x76543210, 0xfedcba98, false, false);
#else
  return __shfl_xor(x, 16, 32);
#endif
}
__device__ __forceinline__ float swap16f(float x) {
  return __builtin_bit_cast(float, swap16i(__builtin_bit_cast(int, x)));
}
__device__ __forceinline__ unsigned swap16u(unsigned x) {
  return (unsigned)swap16i((int)x);
}

// ---------------------------------------------------------------------------
// f32 -> f16 conversion pre-pass (8 elements / thread, nelem % 8 == 0)
// ---------------------------------------------------------------------------
__global__ __launch_bounds__(256)
void cvt_f16(const float* __restrict__ src, _Float16* __restrict__ dst) {
  const size_t i = ((size_t)blockIdx.x * 256 + threadIdx.x) * 8;
  const float4 a = *(const float4*)(src + i);
  const float4 b = *(const float4*)(src + i + 4);
  union { uint4 u; _Float16 h[8]; } r;
  r.h[0] = (_Float16)a.x; r.h[1] = (_Float16)a.y;
  r.h[2] = (_Float16)a.z; r.h[3] = (_Float16)a.w;
  r.h[4] = (_Float16)b.x; r.h[5] = (_Float16)b.y;
  r.h[6] = (_Float16)b.z; r.h[7] = (_Float16)b.w;
  *(uint4*)(dst + i) = r.u;
}

// ---------------------------------------------------------------------------
// GEMM: out[m,n] = (sum_k X[m,k]*W[n,k] + bias[n]) * outScale
// W pre-converted to f16. M = B*L = 8192, N = K = 256.
// One wave handles a 16x64 strip (A-fragment reuse x4).
// TRANS: store out[b][n][tok] (Vt layout) instead of out[m][n].
// ---------------------------------------------------------------------------
template <typename InT, typename OutT, bool TRANS>
__global__ __launch_bounds__(256)
void gemm_proj(const InT* __restrict__ X, const _Float16* __restrict__ W,
               const float* __restrict__ bias, OutT* __restrict__ out,
               float outScale) {
  const int lane = threadIdx.x & 31;
  const int wave = threadIdx.x >> 5;
  const int wg   = blockIdx.x * 8 + wave;   // 2048 waves total
  const int mt   = wg >> 2;                 // 0..511  (M tiles)
  const int nt   = wg & 3;                  // 0..3    (64-col strips)
  const int row0 = mt * 16;
  const int col0 = nt * 64;
  const int g    = lane >> 4;               // lane group (K-half select)
  const int n    = lane & 15;

  v8f acc[4] = {};
  for (int k0 = 0; k0 < DD; k0 += 32) {
    // A fragment (16x32, MxK): lane m = lane&15; halves h -> k = h + 16*g
    V16H a;
    {
      const InT* ap = X + (size_t)(row0 + n) * DD + k0 + 16 * g;
#pragma unroll
      for (int h = 0; h < 8; ++h) a.h[h]     = (_Float16)ap[h];
#pragma unroll
      for (int h = 0; h < 8; ++h) a.h[8 + h] = (_Float16)ap[8 + h];
    }
#pragma unroll
    for (int t = 0; t < 4; ++t) {
      // B fragment (32x16, KxN): lane col = lane&15; halves h -> k = h + 16*g
      // 16 consecutive f16 = one 32B load (32B-aligned).
      const _Float16* wp = W + (size_t)(col0 + t * 16 + n) * DD + k0 + 16 * g;
      v16h bf = *(const v16h*)wp;
      acc[t] = __builtin_amdgcn_wmma_f32_16x16x32_f16(
          false, a.v, false, bf, (short)0, acc[t], false, false);
    }
  }
  // C/D layout: VGPR j, lane l -> M = j + 8*(l>>4), N = l&15
#pragma unroll
  for (int t = 0; t < 4; ++t) {
    const int col = col0 + t * 16 + n;
    const float bval = bias[col];
#pragma unroll
    for (int j = 0; j < 8; ++j) {
      const int m = row0 + j + 8 * g;
      const float val = (acc[t][j] + bval) * outScale;
      if (TRANS) {
        const int b = m >> 11, tok = m & (LL - 1);
        out[((size_t)b * DD + col) * LL + tok] = (OutT)val;
      } else {
        out[(size_t)m * DD + col] = (OutT)val;
      }
    }
  }
}

// ---------------------------------------------------------------------------
// Flash attention: one wave per (b, h, 16-query tile).
// S^T = K x Q^T (one WMMA per 16-key tile; K-dim = HD = 32 exactly).
// Online softmax per lane (lane owns one query column of S^T).
// O^T = V^T x P^T : P^T B-fragment taken from each lane's own P (no swaps needed).
// ---------------------------------------------------------------------------
__global__ __launch_bounds__(128)
void flash_attn(const _Float16* __restrict__ Q, const _Float16* __restrict__ K,
                const _Float16* __restrict__ Vt, _Float16* __restrict__ Y) {
  const int lane = threadIdx.x & 31;
  const int wave = threadIdx.x >> 5;
  const int wg   = blockIdx.x * 4 + wave;   // 4096 waves
  const int qt   = wg & 127;
  const int bh   = wg >> 7;
  const int b    = bh >> 3;
  const int h    = bh & 7;
  const int q0   = qt * 16;
  const int g    = lane >> 4;
  const int n    = lane & 15;

  const size_t qkBase = ((size_t)b * LL) * DD + (size_t)h * HD;   // + row*DD
  const size_t vtBase = ((size_t)b * DD + (size_t)h * HD) * LL;   // + d*LL + tok

  // Q B-fragment (fixed for whole tile): lane col = query q0+n; k = h + 8*((h>>3)+g)
  V16H bq;
  {
    const _Float16* qp = Q + qkBase + (size_t)(q0 + n) * DD + 8 * g;
#pragma unroll
    for (int i = 0; i < 16; ++i) bq.h[i] = qp[i + (i >> 3) * 8];
  }

  float m_run = -1e30f, l_run = 0.0f;
  v8f ot0 = {}, ot1 = {};
  const v8f zero = {};

  const int kend = q0 + 16;                 // keys [0, kend) (causal)
  for (int kb = 0; kb < kend; kb += 32) {
    // --- K A-fragments: tile0 = keys kb..kb+15, tile1 = kb+16..kb+31 ---
    V16H ak0, ak1;
    {
      const _Float16* kp0 = K + qkBase + (size_t)(kb + n) * DD + 8 * g;
      const _Float16* kp1 = kp0 + (size_t)16 * DD;
#pragma unroll
      for (int i = 0; i < 8; ++i) { ak0.h[i] = kp0[i]; ak0.h[8 + i] = kp0[16 + i]; }
#pragma unroll
      for (int i = 0; i < 8; ++i) { ak1.h[i] = kp1[i]; ak1.h[8 + i] = kp1[16 + i]; }
    }
    if (kb + 32 < kend)  // hint next K block toward the caches
      __builtin_prefetch(K + qkBase + (size_t)(kb + 32 + n) * DD, 0, 1);

    // S^T tiles: rows = keys, cols = queries (Q pre-scaled by 1/sqrt(HD))
    v8f s0 = __builtin_amdgcn_wmma_f32_16x16x32_f16(false, ak0.v, false, bq.v,
                                                    (short)0, zero, false, false);
    v8f s1 = __builtin_amdgcn_wmma_f32_16x16x32_f16(false, ak1.v, false, bq.v,
                                                    (short)0, zero, false, false);

    // --- causal mask (only near the diagonal) ---
    if (kb + 31 > q0) {
      const int query = q0 + n;
#pragma unroll
      for (int j = 0; j < 8; ++j) {
        const int key0 = kb + j + 8 * g;
        s0[j] = (key0 > query)      ? -1e30f : s0[j];
        s1[j] = (key0 + 16 > query) ? -1e30f : s1[j];
      }
    }

    // --- online softmax (lane owns query n; partner half-wave has other 16 keys) ---
    float mb = s0[0];
#pragma unroll
    for (int j = 1; j < 8; ++j) mb = fmaxf(mb, s0[j]);
#pragma unroll
    for (int j = 0; j < 8; ++j) mb = fmaxf(mb, s1[j]);
    mb = fmaxf(mb, swap16f(mb));
    const float m_new = fmaxf(m_run, mb);
    const float alpha = __expf(m_run - m_new);

    float p0[8], p1[8], ps = 0.0f;
#pragma unroll
    for (int j = 0; j < 8; ++j) { p0[j] = __expf(s0[j] - m_new); ps += p0[j]; }
#pragma unroll
    for (int j = 0; j < 8; ++j) { p1[j] = __expf(s1[j] - m_new); ps += p1[j]; }
    ps += swap16f(ps);
    l_run = l_run * alpha + ps;
    m_run = m_new;
#pragma unroll
    for (int j = 0; j < 8; ++j) { ot0[j] *= alpha; ot1[j] *= alpha; }

    // --- build B-fragment of P^T (32 keys x 16 queries) from own P (no swaps) ---
    // Own pk0/pk1 already hold keys 8g..8g+7 / 16+8g..16+8g+7 (table order);
    // the swapped value recv is unused.
    unsigned pk0[4], pk1[4];
#pragma unroll
    for (int t = 0; t < 4; ++t) {
      PKH u0; u0.f[0] = (_Float16)p0[2 * t]; u0.f[1] = (_Float16)p0[2 * t + 1];
      PKH u1; u1.f[0] = (_Float16)p1[2 * t]; u1.f[1] = (_Float16)p1[2 * t + 1];
      pk0[t] = u0.u; pk1[t] = u1.u;
    }
    V16H bp;
#pragma unroll
    for (int t = 0; t < 4; ++t) {
      const unsigned send = (g == 0) ? pk1[t] : pk0[t];
      const unsigned recv = swap16u(send);
      bp.u[t]     = pk0[t];   // halves 0..7  : K = 8g..8g+7
      bp.u[4 + t] = pk1[t]; // halves 8..15 : K = 16+8g..16+8g+7
    }

    // --- V^T A-fragments from Vt (contiguous along keys) ---
    V16H av0, av1;
    {
      const _Float16* vp0 = Vt + vtBase + (size_t)n * LL + kb + 8 * g;        // d = n
      const _Float16* vp1 = Vt + vtBase + (size_t)(16 + n) * LL + kb + 8 * g; // d = 16+n
#pragma unroll
      for (int i = 0; i < 8; ++i) { av0.h[i] = vp0[i]; av0.h[8 + i] = vp0[16 + i]; }
#pragma unroll
      for (int i = 0; i < 8; ++i) { av1.h[i] = vp1[i]; av1.h[8 + i] = vp1[16 + i]; }
    }
    ot0 = __builtin_amdgcn_wmma_f32_16x16x32_f16(false, av0.v, false, bp.v,
                                                 (short)0, ot0, false, false);
    ot1 = __builtin_amdgcn_wmma_f32_16x16x32_f16(false, av1.v, false, bp.v,
                                                 (short)0, ot1, false, false);
  }

  // --- normalize and store (lane: query q0+n; O^T rows -> d = j + 8g (+16)) ---
  const float inv = 1.0f / l_run;
  _Float16* yp = Y + qkBase + (size_t)(q0 + n) * DD;
#pragma unroll
  for (int j = 0; j < 8; ++j) {
    yp[j + 8 * g]      = (_Float16)(ot0[j] * inv);
    yp[16 + j + 8 * g] = (_Float16)(ot1[j] * inv);
  }
}

// ---------------------------------------------------------------------------
extern "C" void kernel_launch(void* const* d_in, const int* in_sizes, int n_in,
                              void* d_out, int out_size, void* d_ws, size_t ws_size,
                              hipStream_t stream) {
  const float* key   = (const float*)d_in[0];
  const float* value = (const float*)d_in[1];
  const float* query = (const float*)d_in[2];
  const float* Wk    = (const float*)d_in[3];
  const float* bk    = (const float*)d_in[4];
  const float* Wq    = (const float*)d_in[5];
  const float* bq    = (const float*)d_in[6];
  const float* Wv    = (const float*)d_in[7];
  const float* bv    = (const float*)d_in[8];
  const float* Wp    = (const float*)d_in[9];
  const float* bp    = (const float*)d_in[10];
  float* out = (float*)d_out;

  const size_t N  = (size_t)BB * LL * DD;  // 2M elements per activation buffer
  const size_t WN = (size_t)DD * DD;       // 65536 elements per weight
  _Float16* Qb  = (_Float16*)d_ws;
  _Float16* Kb  = Qb + N;
  _Float16* Vt  = Kb + N;
  _Float16* Yb  = Vt + N;
  _Float16* Whq = Yb + N;
  _Float16* Whk = Whq + WN;
  _Float16* Whv = Whk + WN;
  _Float16* Whp = Whv + WN;

  // weights to f16 once (each reused by 512 M-tiles)
  const int wBlocks = (int)(WN / (8 * 256));  // 32
  cvt_f16<<<wBlocks, 256, 0, stream>>>(Wq, Whq);
  cvt_f16<<<wBlocks, 256, 0, stream>>>(Wk, Whk);
  cvt_f16<<<wBlocks, 256, 0, stream>>>(Wv, Whv);
  cvt_f16<<<wBlocks, 256, 0, stream>>>(Wp, Whp);

  const float qscale = 1.0f / sqrtf((float)HD);

  // Q/K/V projections (f32 -> f16), Q pre-scaled, V stored transposed per batch
  gemm_proj<float, _Float16, false><<<256, 256, 0, stream>>>(query, Whq, bq, Qb, qscale);
  gemm_proj<float, _Float16, false><<<256, 256, 0, stream>>>(key,   Whk, bk, Kb, 1.0f);
  gemm_proj<float, _Float16, true ><<<256, 256, 0, stream>>>(value, Whv, bv, Vt, 1.0f);

  // causal flash attention: 4096 waves, 4 waves/block
  flash_attn<<<1024, 128, 0, stream>>>(Qb, Kb, Vt, Yb);

  // output projection (f16 -> f32) into d_out
  gemm_proj<_Float16, float, false><<<256, 256, 0, stream>>>(Yb, Whp, bp, out, 1.0f);
}
